// Block_58609123721506
// MI455X (gfx1250) — matrix operand order repaired and checked by reading, not compiled, not run on hardware
//
#include <hip/hip_runtime.h>

// ---------------------------------------------------------------------------
// DeltaNet block for MI455X (gfx1250, wave32, WMMA bf16 16x16x32)
// Chunked delta-rule (UT transform): T=2048 serial scan -> 32 serial chunk
// steps of 64x64x64 WMMA matmuls. GEMMs: 128x128 tiles, triple-buffered
// Tensor-Data-Mover staging, one barrier per k-step, ds_load_b128 fragments.
// ---------------------------------------------------------------------------

typedef __bf16 bf16;
typedef __attribute__((ext_vector_type(16))) __bf16 v16bf;
typedef __attribute__((ext_vector_type(8)))  float  v8f;
typedef __attribute__((ext_vector_type(4)))  unsigned int v4u;
typedef __attribute__((ext_vector_type(8)))  int v8i;
typedef __attribute__((ext_vector_type(4)))  int v4i;

#define DEVINL __device__ __forceinline__

static constexpr int Bc = 2, Tc = 2048, Dc = 1024, Hc = 16, Dh = 64;
static constexpr int BT = Bc * Tc;          // 4096 rows
static constexpr int NH = Bc * Hc;          // 32 heads total
static constexpr int CH = 64;               // chunk length
static constexpr int NC = Tc / CH;          // 32 chunks / head

#if defined(__AMDGCN__) && __has_builtin(__builtin_amdgcn_tensor_load_to_lds) && \
    __has_builtin(__builtin_amdgcn_s_wait_tensorcnt)
#define HAVE_TDM 1
#else
#define HAVE_TDM 0
#endif

DEVINL v8f wmma_bf16(v16bf a, v16bf b, v8f c) {
  return __builtin_amdgcn_wmma_f32_16x16x32_bf16(
      /*neg_a=*/false, a, /*neg_b=*/false, b,
      /*c_mod=*/(short)0, c, /*reuse_a=*/false, /*reuse_b=*/false);
}

DEVINL v16bf pack_frag(uint4 lo, uint4 hi) {
  struct P { uint4 a, b; } p;
  p.a = lo; p.b = hi;
  return __builtin_bit_cast(v16bf, p);
}

// A frag (16x32, row-major source): lane L(0..15)=row M=L: v0..3 K0..7,
// v4..7 K16..23 ; lane L+16: K8..15 / K24..31.  Two ds_load_b128 per lane.
DEVINL v16bf frag_a(const bf16* s, int ld) {
  const int lane = threadIdx.x & 31;
  const int m = lane & 15, h8 = (lane >> 4) << 3;
  const bf16* p = s + m * ld + h8;
  return pack_frag(*(const uint4*)p, *(const uint4*)(p + 16));
}
// B frag (32x16) from n-major storage: arr[n*ld + k] == Bop[k][n].
// lane L(0..15)=col N=L holds K0..7,K16..23 ; lane L+16 holds K8..15,K24..31.
DEVINL v16bf frag_bn(const bf16* s, int ld) {
  const int lane = threadIdx.x & 31;
  const int n = lane & 15, h16 = (lane >> 4) << 3;
  const bf16* p = s + n * ld + h16;
  return pack_frag(*(const uint4*)p, *(const uint4*)(p + 16));
}

// ---- 64x64x64 matmul in LDS: C(f32) (+)= A(row-major) @ B (n-major) -------
// 8 waves, 2 tiles of 16x16 per wave, 2 k-steps of 32.
DEVINL void mm64(float* C, const bf16* A, const bf16* Bn, bool accum) {
  const int w = threadIdx.x >> 5;
  const int lane = threadIdx.x & 31;
  const int hv = lane >> 4, nn = lane & 15;
#pragma unroll
  for (int rep = 0; rep < 2; ++rep) {
    const int t = w + rep * 8;
    const int tm = t >> 2, tn = t & 3;
    v8f acc;
#pragma unroll
    for (int r = 0; r < 8; ++r)
      acc[r] = accum ? C[(tm * 16 + hv * 8 + r) * 64 + tn * 16 + nn] : 0.0f;
#pragma unroll
    for (int k0 = 0; k0 < 64; k0 += 32) {
      v16bf fa = frag_a(A + (tm * 16) * 64 + k0, 64);
      v16bf fb = frag_bn(Bn + (tn * 16) * 64 + k0, 64);
      acc = wmma_bf16(fa, fb, acc);
    }
#pragma unroll
    for (int r = 0; r < 8; ++r)
      C[(tm * 16 + hv * 8 + r) * 64 + tn * 16 + nn] = acc[r];
  }
}

#if HAVE_TDM
// Issue a 2D TDM tile load: tile_k x tile_rows (elements, 2B each), source
// row-major with row stride stride_elems, into LDS at lds_addr (row-major).
DEVINL void tdm_load_2d(unsigned lds_addr, const void* gptr,
                        unsigned tile_k, unsigned tile_rows,
                        unsigned stride_elems, unsigned tensor_rows) {
  unsigned long long ga = (unsigned long long)gptr;
  v4u g0;
  g0[0] = 1u;                                   // count=1, user mode
  g0[1] = lds_addr;                             // LDS byte address
  g0[2] = (unsigned)(ga & 0xFFFFFFFFu);         // global_addr[31:0]
  g0[3] = (unsigned)((ga >> 32) & 0x01FFFFFFu) | (2u << 30);  // type=2
  v8i g1;
  g1[0] = (int)(1u << 16);                      // data_size = 2 bytes
  g1[1] = (int)((stride_elems & 0xFFFFu) << 16);        // tensor_dim0 lo
  g1[2] = (int)((stride_elems >> 16) | ((tensor_rows & 0xFFFFu) << 16));
  g1[3] = (int)(((tensor_rows >> 16) & 0xFFFFu) | (tile_k << 16));
  g1[4] = (int)(tile_rows & 0xFFFFu);           // tile_dim1 (tile_dim2=0)
  g1[5] = (int)stride_elems;                    // tensor_dim0_stride lo32
  g1[6] = 0;
  g1[7] = 0;
  v4i g2 = {0, 0, 0, 0};
  v4i g3 = {0, 0, 0, 0};
#if __clang_major__ >= 23
  v8i g4 = {0, 0, 0, 0, 0, 0, 0, 0};
  __builtin_amdgcn_tensor_load_to_lds(g0, g1, g2, g3, g4, 0);
#else
  __builtin_amdgcn_tensor_load_to_lds(g0, g1, g2, g3, 0);
#endif
}
#endif

// ---------------------------------------------------------------------------
__global__ void k_cast(const float* __restrict__ src, bf16* __restrict__ dst, int n) {
  int i = blockIdx.x * 256 + threadIdx.x;
  if (i < n) dst[i] = (bf16)src[i];
}
// transpose-cast: dst[n*K + k] = (bf16)src[k*N + n]   (weights -> N-major)
__global__ void k_cast_t(const float* __restrict__ src, bf16* __restrict__ dst,
                         int K, int N) {
  int i = blockIdx.x * 256 + threadIdx.x;
  if (i >= K * N) return;
  int n = i / K, k = i % K;
  dst[i] = (bf16)src[(size_t)k * N + n];
}

// ---------------------------------------------------------------------------
// WMMA GEMM: C[M,N] f32 = A[M,K] bf16 @ Bt[N,K](n-major) bf16
// BM=128 BN=128 BK=32, 256 threads (8 waves), each wave 32x64 (8 wmma/step).
// Triple-buffered TDM staging, one barrier per k-step.
__global__ void k_gemm(const bf16* __restrict__ A, const bf16* __restrict__ Bt,
                       float* __restrict__ C, int M, int N, int K) {
  const int m0 = blockIdx.y * 128;
  const int n0 = blockIdx.x * 128;
  const int tid = threadIdx.x;
  const int w = tid >> 5, lane = tid & 31;
  const int wm = (w >> 1) * 32;        // wave row offset: 0/32/64/96
  const int wn = (w & 1) * 64;         // wave col offset: 0/64

  v8f acc[2][4];
#pragma unroll
  for (int i = 0; i < 2; ++i)
#pragma unroll
    for (int j = 0; j < 4; ++j)
#pragma unroll
      for (int r = 0; r < 8; ++r) acc[i][j][r] = 0.0f;

#if HAVE_TDM
  __shared__ __attribute__((aligned(16))) bf16 sA[3][128 * 32];
  __shared__ __attribute__((aligned(16))) bf16 sB[3][128 * 32];
  if (tid == 0) {
    tdm_load_2d((unsigned)(unsigned long long)&sA[0][0],
                A + (size_t)m0 * K, 32u, 128u, (unsigned)K, (unsigned)M);
    tdm_load_2d((unsigned)(unsigned long long)&sB[0][0],
                Bt + (size_t)n0 * K, 32u, 128u, (unsigned)K, (unsigned)N);
  }
  int ib = 0;
  for (int kk = 0; kk < K; kk += 32) {
    const int nb = (ib + 1) % 3;
    if (tid == 0) {
      if (kk + 32 < K) {   // prefetch next k-tile while computing this one
        tdm_load_2d((unsigned)(unsigned long long)&sA[nb][0],
                    A + (size_t)m0 * K + kk + 32, 32u, 128u,
                    (unsigned)K, (unsigned)M);
        tdm_load_2d((unsigned)(unsigned long long)&sB[nb][0],
                    Bt + (size_t)n0 * K + kk + 32, 32u, 128u,
                    (unsigned)K, (unsigned)N);
        __builtin_amdgcn_s_wait_tensorcnt(2);   // current buffer complete
      } else {
        __builtin_amdgcn_s_wait_tensorcnt(0);
      }
    }
    __syncthreads();       // buffer ib ready; (3 buffers => WAR-safe w/ 1 bar)
    const bf16* pA = sA[ib];
    const bf16* pB = sB[ib];
    v16bf fa0 = frag_a(pA + (wm + 0) * 32, 32);
    v16bf fa1 = frag_a(pA + (wm + 16) * 32, 32);
    v16bf fb[4];
#pragma unroll
    for (int j = 0; j < 4; ++j) fb[j] = frag_bn(pB + (wn + j * 16) * 32, 32);
#pragma unroll
    for (int j = 0; j < 4; ++j) {
      acc[0][j] = wmma_bf16(fa0, fb[j], acc[0][j]);
      acc[1][j] = wmma_bf16(fa1, fb[j], acc[1][j]);
    }
    ib = nb;
  }
#else
  __shared__ __attribute__((aligned(16))) bf16 sA1[128 * 32];
  __shared__ __attribute__((aligned(16))) bf16 sB1[128 * 32];
  for (int kk = 0; kk < K; kk += 32) {
    for (int i = tid; i < 512; i += 256) {   // 128 rows x 4 uint4 each
      int r = i >> 2, c4 = i & 3;
      ((uint4*)sA1)[i] = *((const uint4*)(A + (size_t)(m0 + r) * K + kk) + c4);
      ((uint4*)sB1)[i] = *((const uint4*)(Bt + (size_t)(n0 + r) * K + kk) + c4);
    }
    if (kk + 32 < K) {
      __builtin_prefetch(A + (size_t)(m0 + (tid >> 1)) * K + kk + 32, 0, 0);
      __builtin_prefetch(Bt + (size_t)(n0 + (tid >> 1)) * K + kk + 32, 0, 0);
    }
    __syncthreads();
    v16bf fa0 = frag_a(sA1 + (wm + 0) * 32, 32);
    v16bf fa1 = frag_a(sA1 + (wm + 16) * 32, 32);
    v16bf fb[4];
#pragma unroll
    for (int j = 0; j < 4; ++j) fb[j] = frag_bn(sB1 + (wn + j * 16) * 32, 32);
#pragma unroll
    for (int j = 0; j < 4; ++j) {
      acc[0][j] = wmma_bf16(fa0, fb[j], acc[0][j]);
      acc[1][j] = wmma_bf16(fa1, fb[j], acc[1][j]);
    }
    __syncthreads();
  }
#endif
  const int hv = lane >> 4, nn = lane & 15;
#pragma unroll
  for (int i = 0; i < 2; ++i)
#pragma unroll
    for (int j = 0; j < 4; ++j)
#pragma unroll
      for (int r = 0; r < 8; ++r)
        C[(size_t)(m0 + wm + i * 16 + hv * 8 + r) * N + n0 + wn + j * 16 + nn] =
            acc[i][j][r];
}

// ---------------------------------------------------------------------------
__global__ void k_beta(const float* __restrict__ x, const float* __restrict__ Wb,
                       float* __restrict__ betap) {
  int gid = blockIdx.x * 256 + threadIdx.x;
  if (gid >= BT * Hc) return;
  int tok = gid >> 4, h = gid & 15;
  const float* xr = x + (size_t)tok * Dc;
  float acc = 0.0f;
  for (int i = 0; i < Dc; ++i) acc += xr[i] * Wb[i * Hc + h];
  betap[gid] = acc;
}

// silu + l2norm on q,k per (token,head); emit bf16 Q, K, betaK, betaV
__global__ void k_qknorm(const float* __restrict__ qf, const float* __restrict__ kf,
                         const float* __restrict__ vf, const float* __restrict__ betap,
                         bf16* __restrict__ q_bf, bf16* __restrict__ k_bf,
                         bf16* __restrict__ kb_bf, bf16* __restrict__ vb_bf) {
  const int blk = blockIdx.x;       // tok*16 + h
  const int tok = blk >> 4, h = blk & 15;
  const int d = threadIdx.x;        // 64 threads
  const size_t idx = (size_t)tok * Dc + h * Dh + d;
  const float beta = 2.0f / (1.0f + __expf(-betap[tok * Hc + h]));
  float xq = qf[idx]; float sq = xq / (1.0f + __expf(-xq));
  float xk = kf[idx]; float sk = xk / (1.0f + __expf(-xk));
  __shared__ float rq[64], rk[64];
  rq[d] = sq * sq; rk[d] = sk * sk;
  __syncthreads();
  for (int s = 32; s > 0; s >>= 1) {
    if (d < s) { rq[d] += rq[d + s]; rk[d] += rk[d + s]; }
    __syncthreads();
  }
  const float iq = rsqrtf(rq[0] + 1e-6f), ik = rsqrtf(rk[0] + 1e-6f);
  const float kn = sk * ik;
  q_bf[idx]  = (bf16)(sq * iq);
  k_bf[idx]  = (bf16)kn;
  kb_bf[idx] = (bf16)(beta * kn);
  vb_bf[idx] = (bf16)(beta * vf[idx]);
}

// ---------------------------------------------------------------------------
// Phase 1 (parallel over head x chunk):
//   A  = strict_tril(betaK @ K^T); Tm = (I+A)^-1 (forward substitution)
//   W  = Tm @ betaK -> bf16 ; U0 = Tm @ betaV -> f32
//   AL = tril(Q @ K^T, incl diag) -> bf16
__global__ void k_phase1(const bf16* __restrict__ q_bf, const bf16* __restrict__ k_bf,
                         const bf16* __restrict__ kb_bf, const bf16* __restrict__ vb_bf,
                         bf16* __restrict__ w_bf, float* __restrict__ u0,
                         bf16* __restrict__ aloc) {
  __shared__ __attribute__((aligned(16))) bf16 Kc[64 * 64], Kb[64 * 64],
      Qc[64 * 64], KbT[64 * 64], VbT[64 * 64], Tb[64 * 64];
  __shared__ __attribute__((aligned(16))) float Af[64 * 64], Cf[64 * 64];

  const int task = blockIdx.x;           // 0..1023
  const int hh = task >> 5, c = task & 31;
  const int b = hh / Hc, h = hh % Hc;
  const int tid = threadIdx.x;           // 256
  const size_t tb = (size_t)(b * Tc + c * CH);

  for (int i = tid; i < 512; i += 256) { // 64 rows x 8 uint4 each
    const int r = i >> 3, c4 = i & 7;
    const size_t g = (tb + r) * Dc + h * Dh;
    uint4 kc4 = *((const uint4*)(k_bf + g) + c4);
    uint4 kb4 = *((const uint4*)(kb_bf + g) + c4);
    uint4 vb4 = *((const uint4*)(vb_bf + g) + c4);
    uint4 qc4 = *((const uint4*)(q_bf + g) + c4);
    ((uint4*)Kc)[i] = kc4;
    ((uint4*)Kb)[i] = kb4;
    ((uint4*)Qc)[i] = qc4;
    const bf16* kbp = (const bf16*)&kb4;
    const bf16* vbp = (const bf16*)&vb4;
#pragma unroll
    for (int e = 0; e < 8; ++e) {        // transposed copies (n-major)
      KbT[(c4 * 8 + e) * 64 + r] = kbp[e];
      VbT[(c4 * 8 + e) * 64 + r] = vbp[e];
    }
  }
  __syncthreads();

  mm64(Af, Kb, Kc, false);               // betaK @ K^T
  __syncthreads();
  for (int i = tid; i < 4096; i += 256) {
    const int r = i >> 6, cc = i & 63;
    Af[i] = (cc < r) ? -Af[i] : 0.0f;    // M = -strict_tril(A)
  }
  __syncthreads();
  for (int i = 1; i < 64; ++i) {         // rows of strict-lower of (I+A)^-1
    float accv = 0.0f;
    if (tid < i) {
      accv = Af[i * 64 + tid];
      for (int j = tid + 1; j < i; ++j) accv += Af[i * 64 + j] * Af[j * 64 + tid];
    }
    __syncthreads();
    if (tid < i) Af[i * 64 + tid] = accv;
    __syncthreads();
  }
  for (int i = tid; i < 4096; i += 256) {
    const int r = i >> 6, cc = i & 63;
    Tb[i] = (bf16)((cc < r) ? Af[i] : ((cc == r) ? 1.0f : 0.0f));
  }
  __syncthreads();

  const size_t ob = (size_t)task * 4096;
  mm64(Cf, Tb, KbT, false);              // W = Tm @ betaK
  __syncthreads();
  for (int i = tid; i < 4096; i += 256) w_bf[ob + i] = (bf16)Cf[i];
  __syncthreads();
  mm64(Cf, Tb, VbT, false);              // U0 = Tm @ betaV
  __syncthreads();
  for (int i = tid; i < 4096; i += 256) u0[ob + i] = Cf[i];
  __syncthreads();
  mm64(Cf, Qc, Kc, false);               // Q @ K^T
  __syncthreads();
  for (int i = tid; i < 4096; i += 256) {
    const int r = i >> 6, cc = i & 63;
    aloc[ob + i] = (bf16)((cc <= r) ? Cf[i] : 0.0f);
  }
}

// ---------------------------------------------------------------------------
// Phase 2 (one block per head, 32 serial chunks, S 64x64 f32 in LDS):
//   U = U0 - W @ S^T ; O = Q @ S^T + AL @ U ; S += U^T @ K ; RMSNorm(O)
__global__ void k_phase2(const bf16* __restrict__ q_bf, const bf16* __restrict__ k_bf,
                         const bf16* __restrict__ w_bf, const float* __restrict__ u0,
                         const bf16* __restrict__ aloc, const float* __restrict__ gamma,
                         bf16* __restrict__ o_bf) {
  __shared__ __attribute__((aligned(16))) float Sf[64 * 64], Uf[64 * 64], Of[64 * 64];
  __shared__ __attribute__((aligned(16))) bf16 Sb[64 * 64], Wc[64 * 64], Qc[64 * 64],
      KcT[64 * 64], Al[64 * 64], UbT[64 * 64];
  __shared__ float red[256], rmsv[64];

  const int hh = blockIdx.x;            // 0..31
  const int b = hh / Hc, h = hh % Hc;
  const int tid = threadIdx.x;          // 256

  for (int i = tid; i < 4096; i += 256) Sf[i] = 0.0f;
  __syncthreads();

  for (int c = 0; c < NC; ++c) {
    const size_t tb = (size_t)(b * Tc + c * CH);
    const size_t ob = (size_t)(hh * NC + c) * 4096;
    for (int i = tid; i < 512; i += 256) {
      const int r = i >> 3, c4 = i & 7;
      const size_t g = (tb + r) * Dc + h * Dh;
      uint4 qc4 = *((const uint4*)(q_bf + g) + c4);
      uint4 kc4 = *((const uint4*)(k_bf + g) + c4);
      ((uint4*)Qc)[i] = qc4;
      ((uint4*)Wc)[i] = ((const uint4*)(w_bf + ob))[i];
      ((uint4*)Al)[i] = ((const uint4*)(aloc + ob))[i];
      const bf16* kcp = (const bf16*)&kc4;
#pragma unroll
      for (int e = 0; e < 8; ++e)        // K chunk stored n-major only
        KcT[(c4 * 8 + e) * 64 + r] = kcp[e];
    }
    for (int i = tid; i < 4096; i += 256) Sb[i] = (bf16)Sf[i];
    __syncthreads();

    mm64(Uf, Wc, Sb, false);             // W @ S^T
    __syncthreads();
    for (int i = tid; i < 4096; i += 256) {
      const float u = u0[ob + i] - Uf[i];
      UbT[(i & 63) * 64 + (i >> 6)] = (bf16)u;   // store U transposed only
    }
    __syncthreads();

    mm64(Of, Qc, Sb, false);             // Q @ S^T
    mm64(Of, Al, UbT, true);             // += tril(QK^T) @ U
    mm64(Sf, UbT, KcT, true);            // S += U^T @ K
    __syncthreads();

    // fused per-head RMSNorm + gamma + bf16 store (token-major)
    const int r = tid >> 2, sg = tid & 3;
    float ss = 0.0f;
    for (int e = 0; e < 16; ++e) {
      const float v = Of[r * 64 + sg * 16 + e];
      ss += v * v;
    }
    red[tid] = ss;
    __syncthreads();
    if (sg == 0)
      rmsv[r] = rsqrtf((red[r * 4] + red[r * 4 + 1] + red[r * 4 + 2] +
                        red[r * 4 + 3]) * (1.0f / 64.0f) + 1e-6f);
    __syncthreads();
    for (int e = 0; e < 16; ++e) {
      const int cc = sg * 16 + e;
      const float v = Of[r * 64 + cc] * rmsv[r] * gamma[h * Dh + cc];
      o_bf[(tb + r) * Dc + h * Dh + cc] = (bf16)v;
    }
    __syncthreads();
  }
}

// ---------------------------------------------------------------------------
__global__ void k_lnorm(const float* __restrict__ ypre, const float* __restrict__ x,
                        const float* __restrict__ g, const float* __restrict__ bb,
                        float* __restrict__ out) {
  __shared__ float sbuf[256];
  const int tok = blockIdx.x, tid = threadIdx.x;
  float v[4]; float sum = 0.0f;
#pragma unroll
  for (int j = 0; j < 4; ++j) {
    const size_t idx = (size_t)tok * Dc + tid + j * 256;
    v[j] = ypre[idx] + x[idx];
    sum += v[j];
  }
  sbuf[tid] = sum; __syncthreads();
  for (int s = 128; s > 0; s >>= 1) { if (tid < s) sbuf[tid] += sbuf[tid + s]; __syncthreads(); }
  const float mu = sbuf[0] * (1.0f / (float)Dc);
  __syncthreads();
  float vs = 0.0f;
#pragma unroll
  for (int j = 0; j < 4; ++j) { const float d = v[j] - mu; vs += d * d; }
  sbuf[tid] = vs; __syncthreads();
  for (int s = 128; s > 0; s >>= 1) { if (tid < s) sbuf[tid] += sbuf[tid + s]; __syncthreads(); }
  const float inv = rsqrtf(sbuf[0] * (1.0f / (float)Dc) + 1e-5f);
#pragma unroll
  for (int j = 0; j < 4; ++j) {
    const int col = tid + j * 256;
    out[(size_t)tok * Dc + col] = (v[j] - mu) * inv * g[col] + bb[col];
  }
}

// ---------------------------------------------------------------------------
extern "C" void kernel_launch(void* const* d_in, const int* in_sizes, int n_in,
                              void* d_out, int out_size, void* d_ws, size_t ws_size,
                              hipStream_t stream) {
  const float* x     = (const float*)d_in[0];
  const float* Wq    = (const float*)d_in[1];
  const float* Wk    = (const float*)d_in[2];
  const float* Wv    = (const float*)d_in[3];
  const float* Wb    = (const float*)d_in[4];
  const float* gam   = (const float*)d_in[5];
  const float* Wo    = (const float*)d_in[6];
  const float* ln_g  = (const float*)d_in[7];
  const float* ln_b  = (const float*)d_in[8];
  float* out = (float*)d_out;

  char* ws = (char*)d_ws;
  size_t off = 0;
  auto alloc = [&](size_t bytes) -> void* {
    void* p = ws + off;
    off += (bytes + 255) & ~(size_t)255;
    return p;
  };
  bf16*  x_bf   = (bf16*)alloc((size_t)BT * Dc * 2);
  bf16*  wqT    = (bf16*)alloc((size_t)Dc * Dc * 2);   // N-major (transposed)
  bf16*  wkT    = (bf16*)alloc((size_t)Dc * Dc * 2);
  bf16*  wvT    = (bf16*)alloc((size_t)Dc * Dc * 2);
  bf16*  woT    = (bf16*)alloc((size_t)Dc * Dc * 2);
  float* q_f    = (float*)alloc((size_t)BT * Dc * 4);
  float* k_f    = (float*)alloc((size_t)BT * Dc * 4);
  float* v_f    = (float*)alloc((size_t)BT * Dc * 4);
  float* betap  = (float*)alloc((size_t)BT * Hc * 4);
  bf16*  q_bf   = (bf16*)alloc((size_t)BT * Dc * 2);
  bf16*  k_bf   = (bf16*)alloc((size_t)BT * Dc * 2);
  bf16*  kb_bf  = (bf16*)alloc((size_t)BT * Dc * 2);
  bf16*  vb_bf  = (bf16*)alloc((size_t)BT * Dc * 2);
  bf16*  w_bf   = (bf16*)alloc((size_t)NH * NC * 4096 * 2);
  float* u0     = (float*)alloc((size_t)NH * NC * 4096 * 4);
  bf16*  aloc_b = (bf16*)alloc((size_t)NH * NC * 4096 * 2);
  bf16*  o_bf   = (bf16*)alloc((size_t)BT * Dc * 2);
  float* ypre   = (float*)alloc((size_t)BT * Dc * 4);
  (void)n_in; (void)in_sizes; (void)out_size; (void)ws_size;

  // casts (weights transposed to N-major for contiguous B-fragments)
  k_cast<<<(BT * Dc + 255) / 256, 256, 0, stream>>>(x, x_bf, BT * Dc);
  k_cast_t<<<(Dc * Dc + 255) / 256, 256, 0, stream>>>(Wq, wqT, Dc, Dc);
  k_cast_t<<<(Dc * Dc + 255) / 256, 256, 0, stream>>>(Wk, wkT, Dc, Dc);
  k_cast_t<<<(Dc * Dc + 255) / 256, 256, 0, stream>>>(Wv, wvT, Dc, Dc);
  k_cast_t<<<(Dc * Dc + 255) / 256, 256, 0, stream>>>(Wo, woT, Dc, Dc);

  // q/k/v projections (WMMA, triple-buffered TDM tiles)
  dim3 gg(Dc / 128, BT / 128);
  k_gemm<<<gg, 256, 0, stream>>>(x_bf, wqT, q_f, BT, Dc, Dc);
  k_gemm<<<gg, 256, 0, stream>>>(x_bf, wkT, k_f, BT, Dc, Dc);
  k_gemm<<<gg, 256, 0, stream>>>(x_bf, wvT, v_f, BT, Dc, Dc);

  // beta logits + silu/l2norm epilogues
  k_beta<<<(BT * Hc + 255) / 256, 256, 0, stream>>>(x, Wb, betap);
  k_qknorm<<<BT * Hc, 64, 0, stream>>>(q_f, k_f, v_f, betap,
                                       q_bf, k_bf, kb_bf, vb_bf);

  // chunked delta rule
  k_phase1<<<NH * NC, 256, 0, stream>>>(q_bf, k_bf, kb_bf, vb_bf,
                                        w_bf, u0, aloc_b);
  k_phase2<<<NH, 256, 0, stream>>>(q_bf, k_bf, w_bf, u0, aloc_b, gam, o_bf);

  // output projection + residual LayerNorm
  k_gemm<<<gg, 256, 0, stream>>>(o_bf, woT, ypre, BT, Dc, Dc);
  k_lnorm<<<BT, 256, 0, stream>>>(ypre, x, ln_g, ln_b, out);
}
